// MOELayer_12910671692088
// MI455X (gfx1250) — matrix operand order repaired and checked by reading, not compiled, not run on hardware
//
#include <hip/hip_runtime.h>

#define NUM_EXPERT 16
#define IN_FEAT    1024
#define OUT_FEAT   2048
#define N_TOKENS   4096

#define TILE_M 128
#define TILE_N 128
#define TILE_K 32
#define LDS_PITCH 40   // shorts per row: 32 data + 8 pad -> 80B rows, 16B aligned

typedef __attribute__((ext_vector_type(16))) __bf16         v16bf;
typedef __attribute__((ext_vector_type(8)))  float          v8f;
typedef __attribute__((ext_vector_type(4)))  float          v4f;
typedef __attribute__((ext_vector_type(8)))  unsigned short v8u16;
typedef __attribute__((ext_vector_type(16))) unsigned short v16u16;
typedef __attribute__((ext_vector_type(4)))  unsigned       v4u;

// Pack two f32 -> two bf16 in one dword (low = f0, high = f1).
// Prefer the HW packed convert (RNE); fallback: round-half-up + one v_perm_b32.
__device__ __forceinline__ unsigned packbf2(float f0, float f1) {
#if __has_builtin(__builtin_amdgcn_cvt_pk_bf16_f32)
  typedef __attribute__((ext_vector_type(2))) __bf16 v2bf;
  v2bf p = __builtin_amdgcn_cvt_pk_bf16_f32(f0, f1);
  return __builtin_bit_cast(unsigned, p);
#else
  unsigned u0 = __builtin_bit_cast(unsigned, f0) + 0x8000u;
  unsigned u1 = __builtin_bit_cast(unsigned, f1) + 0x8000u;
  return __builtin_amdgcn_perm(u1, u0, 0x07060302u); // {u1.hi16, u0.hi16}
#endif
}

__device__ __forceinline__ v8u16 pack8(v4f x0, v4f x1) {
  v4u p;
  p[0] = packbf2(x0[0], x0[1]);
  p[1] = packbf2(x0[2], x0[3]);
  p[2] = packbf2(x1[0], x1[1]);
  p[3] = packbf2(x1[2], x1[3]);
  return __builtin_bit_cast(v8u16, p);
}

__device__ __forceinline__ v16bf frag16(v8u16 lo, v8u16 hi) {
  v16u16 t = __builtin_shufflevector(lo, hi, 0,1,2,3,4,5,6,7,8,9,10,11,12,13,14,15);
  return __builtin_bit_cast(v16bf, t);
}

// ---------------- routing: build per-expert token lists ----------------

__global__ void moe_zero_counts(int* cnt) {
  if (threadIdx.x < NUM_EXPERT) cnt[threadIdx.x] = 0;
}

__global__ void moe_route(const int* __restrict__ gate, int* __restrict__ cnt,
                          int* __restrict__ list) {
  int b = blockIdx.x * blockDim.x + threadIdx.x;
  if (b < N_TOKENS) {
    int e = gate[b];
    int pos = atomicAdd(&cnt[e], 1);
    list[e * N_TOKENS + pos] = b;
  }
}

// ---------------- expert GEMM: out[tok] = W[e] @ x[tok] ----------------
// Block: 256 threads = 8 waves. Tile: 128 tokens x 128 out-features.
// Wave (wm in 0..1, wn in 0..3) computes a 64x32 sub-tile as 4x2 WMMA accums.

__launch_bounds__(256)
__global__ void moe_gemm(const float* __restrict__ x,
                         const float* __restrict__ weight,
                         const int*   __restrict__ cnt,
                         const int*   __restrict__ list,
                         float*       __restrict__ out,
                         float*       __restrict__ dump) {
  __shared__ alignas(16) unsigned short As[TILE_M * LDS_PITCH];
  __shared__ alignas(16) unsigned short Bs[TILE_N * LDS_PITCH];
  __shared__ alignas(16) int s_tok[TILE_M];

  const int n_ttiles = N_TOKENS / TILE_M;   // 32
  const int n_ntiles = OUT_FEAT / TILE_N;   // 16

  int bid    = blockIdx.x;
  int e      = bid / (n_ttiles * n_ntiles);
  int rem    = bid % (n_ttiles * n_ntiles);
  int tile_t = rem / n_ntiles;
  int tile_n = rem % n_ntiles;

  int count = cnt[e];
  int m0 = tile_t * TILE_M;
  if (m0 >= count) return;   // empty token tile for this expert

  int tid = threadIdx.x;

  if (tid < TILE_M) {
    int r = m0 + tid;
    s_tok[tid] = (r < count) ? list[e * N_TOKENS + r] : 0; // clamp: safe loads
  }
  __syncthreads();

  const float* wbase = weight + (size_t)e * OUT_FEAT * IN_FEAT
                              + (size_t)tile_n * TILE_N * IN_FEAT;

  const int wave = tid >> 5;
  const int lane = tid & 31;
  const int wm   = wave & 1;   // M sub-block (64 rows each)
  const int wn   = wave >> 1;  // N sub-block (32 cols each)
  const int kh   = lane >> 4;  // K-half select (this code's contiguous k order)
  const int ml   = lane & 15;

  // Loader mapping: thread t handles row t>>1, 16 floats starting at (t&1)*16.
  const int lr = tid >> 1;
  const int lc = (tid & 1) * 16;
  const int tokr = s_tok[lr];
  const float* arow = x + (size_t)tokr * IN_FEAT + lc;
  const float* brow = wbase + (size_t)lr * IN_FEAT + lc;
  unsigned short* a_st = As + lr * LDS_PITCH + lc;
  unsigned short* b_st = Bs + lr * LDS_PITCH + lc;

  v8f acc[4][2] = {};

  for (int k0 = 0; k0 < IN_FEAT; k0 += TILE_K) {
    // Stage fp32 -> registers (b128 loads)
    v4f a0 = *(const v4f*)(arow + k0 +  0);
    v4f a1 = *(const v4f*)(arow + k0 +  4);
    v4f a2 = *(const v4f*)(arow + k0 +  8);
    v4f a3 = *(const v4f*)(arow + k0 + 12);
    v4f b0 = *(const v4f*)(brow + k0 +  0);
    v4f b1 = *(const v4f*)(brow + k0 +  4);
    v4f b2 = *(const v4f*)(brow + k0 +  8);
    v4f b3 = *(const v4f*)(brow + k0 + 12);

    if (k0 + TILE_K < IN_FEAT) {   // hint the streaming pattern (global_prefetch_b8)
      __builtin_prefetch(brow + k0 + TILE_K, 0, 1);
      __builtin_prefetch(arow + k0 + TILE_K, 0, 1);
    }

    __syncthreads();  // previous tile fully consumed
    *(v8u16*)(a_st)     = pack8(a0, a1);
    *(v8u16*)(a_st + 8) = pack8(a2, a3);
    *(v8u16*)(b_st)     = pack8(b0, b1);
    *(v8u16*)(b_st + 8) = pack8(b2, b3);
    __syncthreads();  // tile visible

    // B fragments: lane -> col (ml), elements 0..15 = K kh*16 .. kh*16+15
    v16bf bf[2];
#pragma unroll
    for (int ni = 0; ni < 2; ++ni) {
      const unsigned short* bp = Bs + (wn * 32 + ni * 16 + ml) * LDS_PITCH + kh * 16;
      bf[ni] = frag16(*(const v8u16*)bp, *(const v8u16*)(bp + 8));
    }

    // A fragments: lane -> row (ml), elements 0..15 = K kh*16 .. kh*16+15
#pragma unroll
    for (int mi = 0; mi < 4; ++mi) {
      const unsigned short* ap = As + (wm * 64 + mi * 16 + ml) * LDS_PITCH + kh * 16;
      v16bf af = frag16(*(const v8u16*)ap, *(const v8u16*)(ap + 8));
#pragma unroll
      for (int ni = 0; ni < 2; ++ni) {
        acc[mi][ni] = __builtin_amdgcn_wmma_f32_16x16x32_bf16(
            false, af, false, bf[ni], (short)0, acc[mi][ni], false, false);
      }
    }
  }

  // Store: D layout -> lane l, vreg v holds (row = (l>>4)*8 + v, col = l&15).
  // Branch-free: invalid rows write to the scratch dump row instead of out.
  const int col0 = tile_n * TILE_N + wn * 32 + ml;
#pragma unroll
  for (int mi = 0; mi < 4; ++mi) {
    const int rbase = wm * 64 + mi * 16 + kh * 8;
    int toks[8];
#pragma unroll
    for (int v = 0; v < 8; ++v) toks[v] = s_tok[rbase + v];
#pragma unroll
    for (int v = 0; v < 8; ++v) {
      float* rowp = (m0 + rbase + v < count)
                        ? out + (size_t)toks[v] * OUT_FEAT
                        : dump;
      rowp[col0]      = acc[mi][0][v];
      rowp[col0 + 16] = acc[mi][1][v];
    }
  }
}

// ---------------- launch ----------------

extern "C" void kernel_launch(void* const* d_in, const int* in_sizes, int n_in,
                              void* d_out, int out_size, void* d_ws, size_t ws_size,
                              hipStream_t stream) {
  const float* x    = (const float*)d_in[0];
  const int*   gate = (const int*)d_in[1];
  const float* w    = (const float*)d_in[2];
  float*       out  = (float*)d_out;

  int*   cnt  = (int*)d_ws;               // 16 counters (padded to 64 ints)
  int*   list = cnt + 64;                 // 16 * 4096 token indices
  float* dump = (float*)(list + NUM_EXPERT * N_TOKENS);  // OUT_FEAT scratch floats

  moe_zero_counts<<<1, NUM_EXPERT, 0, stream>>>(cnt);
  moe_route<<<N_TOKENS / 256, 256, 0, stream>>>(gate, cnt, list);

  int grid = NUM_EXPERT * (N_TOKENS / TILE_M) * (OUT_FEAT / TILE_N);
  moe_gemm<<<grid, 256, 0, stream>>>(x, w, cnt, list, out, dump);
}
